// Model_17119739641981
// MI455X (gfx1250) — matrix operand order repaired and checked by reading, not compiled, not run on hardware
//
#include <hip/hip_runtime.h>
#include <math.h>
#include <stddef.h>

// ---------------- model constants ----------------
#define BB   2
#define LL   1024
#define TT   (BB*LL)        // 2048 tokens
#define DD   1024
#define NL   4
#define NH   16
#define NKV  4
#define REP  (NH/NKV)
#define DH   64             // head dim
#define DFF  4096
#define VV   32000
#define EPS  1e-5f
#define LOG2_THETA 13.287712379549449f   // log2(10000)

typedef __bf16 bf16;
typedef __attribute__((ext_vector_type(16))) __bf16 v16bf;
typedef __attribute__((ext_vector_type(8)))  __bf16 v8bf;
typedef __attribute__((ext_vector_type(8)))  float  v8f;

typedef __attribute__((ext_vector_type(4))) unsigned int u32x4;
typedef __attribute__((ext_vector_type(8))) int          i32x8;
typedef __attribute__((ext_vector_type(4))) int          i32x4;

__device__ __forceinline__ bf16 f2bf(float f) {
    union { float f; unsigned u; } v; v.f = f;
    unsigned r = v.u + 0x7FFFu + ((v.u >> 16) & 1u);   // round-to-nearest-even
    unsigned short h = (unsigned short)(r >> 16);
    union { unsigned short s; bf16 b; } o; o.s = h;
    return o.b;
}

__device__ __forceinline__ v8f wmma_bf16(v16bf a, v16bf b, v8f c) {
    return __builtin_amdgcn_wmma_f32_16x16x32_bf16(false, a, false, b, (short)0, c,
                                                   false, false);
}

__device__ __forceinline__ v16bf cat8(v8bf lo, v8bf hi) {
    return __builtin_shufflevector(lo, hi, 0, 1, 2, 3, 4, 5, 6, 7,
                                          8, 9, 10, 11, 12, 13, 14, 15);
}

// ---------------- Tensor Data Mover tile load ----------------
// Moves a rows x 32 bf16 tile (row stride K elems) from global into LDS,
// padding 4 DWORDs after every 16 DWORDs -> LDS rows of 40 bf16 (80 B).
#define BM 128
#define BN 128
#define BKK 32
#define LDP (BKK + 8)

__device__ __forceinline__ void tdm_load_tile(const bf16* gsrc, void* ldsdst,
                                              int K, int rows) {
    unsigned long long gaddr = (unsigned long long)(size_t)gsrc;
    unsigned ldsoff = (unsigned)(size_t)ldsdst;       // low 32 bits = LDS offset
    u32x4 g0;
    g0[0] = 1u;                                       // count = 1 valid descriptor
    g0[1] = ldsoff;                                   // lds_addr
    g0[2] = (unsigned)(gaddr & 0xFFFFFFFFu);          // global_addr lo
    g0[3] = (unsigned)((gaddr >> 32) & 0x01FFFFFFu) | (2u << 30);  // addr hi | type=2
    unsigned long long q0 =
        (1ull << 16)                                  // data_size = 2 bytes
      | (1ull << 20)                                  // pad_enable
      | (3ull << 22)                                  // pad_interval = 16 DWORDs
      | (3ull << 25)                                  // pad_amount  = 4 DWORDs
      | ((unsigned long long)(K & 0xFFFF) << 48);     // tensor_dim0 lo
    unsigned long long q1 =
        ((unsigned long long)(K >> 16) & 0xFFFFull)   // tensor_dim0 hi
      | ((unsigned long long)(unsigned)rows << 16)    // tensor_dim1
      | ((unsigned long long)BKK << 48);              // tile_dim0 = 32
    unsigned long long q2 =
        (unsigned long long)(unsigned)rows            // tile_dim1
      | ((unsigned long long)(unsigned)K << 32);      // tensor_dim0_stride lo
    unsigned long long q3 = 0;                        // stride hi, dim1_stride
    i32x8 g1;
    g1[0] = (int)q0; g1[1] = (int)(q0 >> 32);
    g1[2] = (int)q1; g1[3] = (int)(q1 >> 32);
    g1[4] = (int)q2; g1[5] = (int)(q2 >> 32);
    g1[6] = (int)q3; g1[7] = (int)(q3 >> 32);
    i32x4 z4 = {0, 0, 0, 0};
#if defined(__clang_major__) && __clang_major__ >= 23
    i32x8 z8 = {0, 0, 0, 0, 0, 0, 0, 0};
    __builtin_amdgcn_tensor_load_to_lds(g0, g1, z4, z4, z8, 0);
#else
    __builtin_amdgcn_tensor_load_to_lds(g0, g1, z4, z4, 0);
#endif
}

// ---------------- elementwise kernels ----------------
// transpose-cast: src [K][N] f32  ->  dst [N][K] bf16
__global__ __launch_bounds__(256)
void cast_transpose_kernel(const float* __restrict__ src, bf16* __restrict__ dst,
                           int Kd, int Nd) {
    __shared__ float tile[32][33];
    int kb = blockIdx.y * 32, nb = blockIdx.x * 32;
    int tx = threadIdx.x, ty = threadIdx.y;      // 32 x 8
    for (int i = 0; i < 32; i += 8)
        tile[ty + i][tx] = src[(size_t)(kb + ty + i) * Nd + nb + tx];
    __syncthreads();
    for (int i = 0; i < 32; i += 8)
        dst[(size_t)(nb + ty + i) * Kd + kb + tx] = f2bf(tile[tx][ty + i]);
}

__global__ void embed_kernel(const int* __restrict__ ti,
                             const float* __restrict__ emb,
                             float* __restrict__ x) {
    int tok = blockIdx.x;
    size_t row = (size_t)ti[tok];
    for (int i = threadIdx.x; i < DD; i += blockDim.x)
        x[(size_t)tok * DD + i] = emb[row * DD + i];
}

__global__ __launch_bounds__(256)
void rmsnorm_kernel(const float* __restrict__ x, const float* __restrict__ w,
                    bf16* __restrict__ out) {
    int row = blockIdx.x;
    int t = threadIdx.x;
    __shared__ float red[256];
    float p = 0.f;
    for (int c = t; c < DD; c += 256) {
        float v = x[(size_t)row * DD + c];
        p += v * v;
    }
    red[t] = p;
    __syncthreads();
    for (int s = 128; s > 0; s >>= 1) {
        if (t < s) red[t] += red[t + s];
        __syncthreads();
    }
    float rs = rsqrtf(red[0] / (float)DD + EPS);
    for (int c = t; c < DD; c += 256)
        out[(size_t)row * DD + c] = f2bf(x[(size_t)row * DD + c] * rs * w[c]);
}

__global__ void silu_mul_kernel(const float* __restrict__ g,
                                const float* __restrict__ u,
                                bf16* __restrict__ out, size_t n) {
    size_t i = (size_t)blockIdx.x * blockDim.x + threadIdx.x;
    if (i < n) {
        float gv = g[i];
        float s = gv / (1.f + __expf(-gv));
        out[i] = f2bf(s * u[i]);
    }
}

// RoPE + cast. q -> [B,NH,L,DH], k -> [B,NKV,L,DH], v -> [B,NKV,DH,L] (transposed)
__global__ __launch_bounds__(256)
void rope_cast_kernel(const float* __restrict__ qf, const float* __restrict__ kf,
                      const float* __restrict__ vf,
                      bf16* __restrict__ qb, bf16* __restrict__ kb,
                      bf16* __restrict__ vb) {
    int tok = blockIdx.x;
    int b = tok / LL, l = tok % LL;
    int t = threadIdx.x;
    for (int pid = t; pid < NH * 32; pid += 256) {
        int head = pid >> 5, pi = pid & 31;
        float th = exp2f(-(float)pi * (LOG2_THETA / 32.f));
        float ang = (float)l * th;
        float c = cosf(ang), s = sinf(ang);
        float r  = qf[(size_t)tok * DD + head * DH + 2 * pi];
        float im = qf[(size_t)tok * DD + head * DH + 2 * pi + 1];
        size_t dst = ((size_t)(b * NH + head) * LL + l) * DH + 2 * pi;
        qb[dst]     = f2bf(r * c - im * s);
        qb[dst + 1] = f2bf(r * s + im * c);
    }
    for (int pid = t; pid < NKV * 32; pid += 256) {
        int head = pid >> 5, pi = pid & 31;
        float th = exp2f(-(float)pi * (LOG2_THETA / 32.f));
        float ang = (float)l * th;
        float c = cosf(ang), s = sinf(ang);
        float r  = kf[(size_t)tok * (NKV * DH) + head * DH + 2 * pi];
        float im = kf[(size_t)tok * (NKV * DH) + head * DH + 2 * pi + 1];
        size_t dst = ((size_t)(b * NKV + head) * LL + l) * DH + 2 * pi;
        kb[dst]     = f2bf(r * c - im * s);
        kb[dst + 1] = f2bf(r * s + im * c);
    }
    for (int c0 = t; c0 < NKV * DH; c0 += 256) {
        int head = c0 >> 6, d = c0 & 63;
        vb[((size_t)(b * NKV + head) * DH + d) * LL + l] =
            f2bf(vf[(size_t)tok * (NKV * DH) + c0]);
    }
}

// ---------------- WMMA GEMM (double buffered, manual staging) ----------------
// C[M,N] f32 = A[M,K] bf16 (row-major) x Bt[N,K] bf16 (pre-transposed) [+Res]
// 256 thr = 8 waves (2x4); block tile 128x128; wave tile 64x32; K-step 32.
__global__ __launch_bounds__(256)
void gemm_bf16_kernel(const bf16* __restrict__ A, const bf16* __restrict__ Bt,
                      float* __restrict__ C, const float* __restrict__ Res,
                      int M, int N, int K) {
    __shared__ bf16 As[2][BM][LDP];
    __shared__ bf16 Bs[2][BN][LDP];
    int t = threadIdx.x;
    int wave = t >> 5, lane = t & 31;
    int wm = wave >> 2, wn = wave & 3;            // 2 x 4 wave grid
    int lrow = lane & 15, kh = lane >> 4;
    int m0 = blockIdx.y * BM, n0 = blockIdx.x * BN;

    int r0 = t >> 2, c0 = (t & 3) * 8;
    int r1 = r0 + 64;

    const bf16* Aw = A + (size_t)m0 * K;
    const bf16* Bw = Bt + (size_t)n0 * K;

    v8f acc[4][2];
    for (int mi = 0; mi < 4; ++mi)
        for (int ni = 0; ni < 2; ++ni)
            for (int v = 0; v < 8; ++v) acc[mi][ni][v] = 0.f;

    *(v8bf*)&As[0][r0][c0] = *(const v8bf*)&Aw[(size_t)r0 * K + c0];
    *(v8bf*)&As[0][r1][c0] = *(const v8bf*)&Aw[(size_t)r1 * K + c0];
    *(v8bf*)&Bs[0][r0][c0] = *(const v8bf*)&Bw[(size_t)r0 * K + c0];
    *(v8bf*)&Bs[0][r1][c0] = *(const v8bf*)&Bw[(size_t)r1 * K + c0];

    int nsteps = K / BKK;
    int buf = 0;
    v8bf ra0, ra1, rb0, rb1;

    for (int step = 0; step < nsteps; ++step) {
        if (step + 1 < nsteps) {
            int kn = (step + 1) * BKK;
            ra0 = *(const v8bf*)&Aw[(size_t)r0 * K + kn + c0];
            ra1 = *(const v8bf*)&Aw[(size_t)r1 * K + kn + c0];
            rb0 = *(const v8bf*)&Bw[(size_t)r0 * K + kn + c0];
            rb1 = *(const v8bf*)&Bw[(size_t)r1 * K + kn + c0];
            if (step + 2 < nsteps) {
                int kp = kn + BKK;
                __builtin_prefetch(&Aw[(size_t)r0 * K + kp + c0], 0, 0);
                __builtin_prefetch(&Bw[(size_t)r0 * K + kp + c0], 0, 0);
            }
        }
        __syncthreads();

        v16bf af[4], bfrag[2];
        for (int mi = 0; mi < 4; ++mi) {
            const bf16* rp = &As[buf][wm * 64 + mi * 16 + lrow][0];
            af[mi] = cat8(*(const v8bf*)(rp + kh * 16),
                          *(const v8bf*)(rp + kh * 16 + 8));
        }
        for (int ni = 0; ni < 2; ++ni) {
            const bf16* rp = &Bs[buf][wn * 32 + ni * 16 + lrow][0];
            bfrag[ni] = cat8(*(const v8bf*)(rp + kh * 16),
                             *(const v8bf*)(rp + kh * 16 + 8));
        }
        for (int mi = 0; mi < 4; ++mi)
            for (int ni = 0; ni < 2; ++ni)
                acc[mi][ni] = wmma_bf16(af[mi], bfrag[ni], acc[mi][ni]);

        if (step + 1 < nsteps) {
            int nb = buf ^ 1;
            *(v8bf*)&As[nb][r0][c0] = ra0;
            *(v8bf*)&As[nb][r1][c0] = ra1;
            *(v8bf*)&Bs[nb][r0][c0] = rb0;
            *(v8bf*)&Bs[nb][r1][c0] = rb1;
        }
        buf ^= 1;
    }

    // epilogue: hoist uniform residual branch
    if (Res) {
        for (int mi = 0; mi < 4; ++mi)
            for (int ni = 0; ni < 2; ++ni)
                for (int v = 0; v < 8; ++v) {
                    int gm = m0 + wm * 64 + mi * 16 + v + 8 * kh;
                    int gn = n0 + wn * 32 + ni * 16 + lrow;
                    C[(size_t)gm * N + gn] = acc[mi][ni][v] + Res[(size_t)gm * N + gn];
                }
    } else {
        for (int mi = 0; mi < 4; ++mi)
            for (int ni = 0; ni < 2; ++ni)
                for (int v = 0; v < 8; ++v) {
                    int gm = m0 + wm * 64 + mi * 16 + v + 8 * kh;
                    int gn = n0 + wn * 32 + ni * 16 + lrow;
                    C[(size_t)gm * N + gn] = acc[mi][ni][v];
                }
    }
}

// ---------------- WMMA GEMM, all tiles staged by the Tensor Data Mover -------
// Double-buffered: TDM fills buffer p^1 while waves compute from buffer p.
// Wave 0 drives the DMA and the TENSORcnt waits; used for the logits GEMM.
__global__ __launch_bounds__(256)
void gemm_bf16_tdm_kernel(const bf16* __restrict__ A, const bf16* __restrict__ Bt,
                          float* __restrict__ C, int M, int N, int K) {
    __shared__ bf16 As[2][BM][LDP];
    __shared__ bf16 Bs[2][BN][LDP];
    int t = threadIdx.x;
    int wave = t >> 5, lane = t & 31;
    int wm = wave >> 2, wn = wave & 3;
    int lrow = lane & 15, kh = lane >> 4;
    int m0 = blockIdx.y * BM, n0 = blockIdx.x * BN;

    const bf16* Aw = A + (size_t)m0 * K;
    const bf16* Bw = Bt + (size_t)n0 * K;

    v8f acc[4][2];
    for (int mi = 0; mi < 4; ++mi)
        for (int ni = 0; ni < 2; ++ni)
            for (int v = 0; v < 8; ++v) acc[mi][ni][v] = 0.f;

    int nsteps = K / BKK;
    if (wave == 0) {                      // kick off tile 0 into buffer 0
        tdm_load_tile(Aw, &As[0][0][0], K, BM);
        tdm_load_tile(Bw, &Bs[0][0][0], K, BN);
    }

    int buf = 0;
    for (int step = 0; step < nsteps; ++step) {
        if (wave == 0)
            __builtin_amdgcn_s_wait_tensorcnt(0);   // tile `step` landed in LDS
        __syncthreads();                            // visible to all; prev reads done
        if (wave == 0 && step + 1 < nsteps) {       // overlap DMA of next tile
            int kn = (step + 1) * BKK;
            tdm_load_tile(Aw + kn, &As[buf ^ 1][0][0], K, BM);
            tdm_load_tile(Bw + kn, &Bs[buf ^ 1][0][0], K, BN);
        }

        v16bf af[4], bfrag[2];
        for (int mi = 0; mi < 4; ++mi) {
            const bf16* rp = &As[buf][wm * 64 + mi * 16 + lrow][0];
            af[mi] = cat8(*(const v8bf*)(rp + kh * 16),
                          *(const v8bf*)(rp + kh * 16 + 8));
        }
        for (int ni = 0; ni < 2; ++ni) {
            const bf16* rp = &Bs[buf][wn * 32 + ni * 16 + lrow][0];
            bfrag[ni] = cat8(*(const v8bf*)(rp + kh * 16),
                             *(const v8bf*)(rp + kh * 16 + 8));
        }
        for (int mi = 0; mi < 4; ++mi)
            for (int ni = 0; ni < 2; ++ni)
                acc[mi][ni] = wmma_bf16(af[mi], bfrag[ni], acc[mi][ni]);
        buf ^= 1;
    }

    for (int mi = 0; mi < 4; ++mi)
        for (int ni = 0; ni < 2; ++ni)
            for (int v = 0; v < 8; ++v) {
                int gm = m0 + wm * 64 + mi * 16 + v + 8 * kh;
                int gn = n0 + wn * 32 + ni * 16 + lrow;
                __builtin_nontemporal_store(acc[mi][ni][v], &C[(size_t)gm * N + gn]);
            }
}

// ---------------- flash attention (WMMA) ----------------
// one wave per (b, head, 16-query tile). q,k: [B,H,L,DH]; v: [B,Hkv,DH,L].
__global__ __launch_bounds__(32)
void attn_kernel(const bf16* __restrict__ Q, const bf16* __restrict__ K,
                 const bf16* __restrict__ Vv, bf16* __restrict__ O) {
    int bh = blockIdx.x;
    int b = bh / NH, h = bh % NH;
    int hkv = h / REP;
    int q0 = blockIdx.y * 16;
    int lane = threadIdx.x & 31;
    int lrow = lane & 15, kh = lane >> 4;

    const bf16* Qh = Q + ((size_t)(b * NH + h) * LL + q0) * DH;
    const bf16* Kh = K + ((size_t)(b * NKV + hkv) * LL) * DH;
    const bf16* Vh = Vv + (size_t)(b * NKV + hkv) * DH * LL;

    v16bf aQ[2];
    for (int ds = 0; ds < 2; ++ds) {
        const bf16* qp = Qh + (size_t)lrow * DH + ds * 32;
        aQ[ds] = cat8(*(const v8bf*)(qp + kh * 16),
                      *(const v8bf*)(qp + kh * 16 + 8));
    }

    v8f o[4];
    for (int nt = 0; nt < 4; ++nt)
        for (int v = 0; v < 8; ++v) o[nt][v] = 0.f;
    float rm[8], rl[8];
    for (int v = 0; v < 8; ++v) { rm[v] = -1e30f; rl[v] = 0.f; }

    __shared__ bf16 Plds[16][32];
    const float scale = 0.125f;   // 1/sqrt(64)

    for (int kbase = 0; kbase <= q0 + 15; kbase += 32) {
        v8f s[2];
        for (int ni = 0; ni < 2; ++ni)
            for (int v = 0; v < 8; ++v) s[ni][v] = 0.f;
        for (int ni = 0; ni < 2; ++ni) {
            int key = kbase + ni * 16 + lrow;
            for (int ds = 0; ds < 2; ++ds) {
                const bf16* kp = Kh + (size_t)key * DH + ds * 32 + kh * 16;
                v16bf bk = cat8(*(const v8bf*)kp, *(const v8bf*)(kp + 8));
                s[ni] = wmma_bf16(aQ[ds], bk, s[ni]);
            }
        }
        float mnew[8];
        for (int v = 0; v < 8; ++v) {
            int grow = q0 + v + 8 * kh;
            float mx = rm[v];
            for (int ni = 0; ni < 2; ++ni) {
                int n = kbase + ni * 16 + lrow;
                float val = s[ni][v] * scale;
                if (n > grow) val = -1e30f;
                s[ni][v] = val;
                mx = fmaxf(mx, val);
            }
            for (int off = 1; off < 16; off <<= 1)
                mx = fmaxf(mx, __shfl_xor(mx, off, 32));
            mnew[v] = mx;
        }
        for (int v = 0; v < 8; ++v) {
            float alpha = __expf(rm[v] - mnew[v]);
            float sum = 0.f;
            for (int ni = 0; ni < 2; ++ni) {
                float p = __expf(s[ni][v] - mnew[v]);
                sum += p;
                Plds[v + 8 * kh][ni * 16 + lrow] = f2bf(p);
            }
            for (int off = 1; off < 16; off <<= 1)
                sum += __shfl_xor(sum, off, 32);
            rl[v] = rl[v] * alpha + sum;
            for (int nt = 0; nt < 4; ++nt) o[nt][v] *= alpha;
            rm[v] = mnew[v];
        }
        __syncthreads();
        const bf16* pp = &Plds[lrow][0];
        v16bf pa = cat8(*(const v8bf*)(pp + kh * 16),
                        *(const v8bf*)(pp + kh * 16 + 8));
        for (int nt = 0; nt < 4; ++nt) {
            const bf16* vc = Vh + (size_t)(nt * 16 + lrow) * LL + kbase + kh * 16;
            v16bf bv = cat8(*(const v8bf*)vc, *(const v8bf*)(vc + 8));
            o[nt] = wmma_bf16(pa, bv, o[nt]);
        }
        __syncthreads();
    }
    for (int v = 0; v < 8; ++v) {
        float inv = 1.f / rl[v];
        int tok = b * LL + q0 + v + 8 * kh;
        for (int nt = 0; nt < 4; ++nt)
            O[(size_t)tok * DD + h * DH + nt * 16 + lrow] = f2bf(o[nt][v] * inv);
    }
}

// ---------------- host orchestration ----------------
static inline size_t alignup(size_t x) { return (x + 255) & ~(size_t)255; }

extern "C" void kernel_launch(void* const* d_in, const int* in_sizes, int n_in,
                              void* d_out, int out_size, void* d_ws, size_t ws_size,
                              hipStream_t stream) {
    (void)in_sizes; (void)n_in; (void)out_size; (void)ws_size;
    const int*   ti    = (const int*)d_in[0];
    const float* emb   = (const float*)d_in[1];
    const float* ln1_w = (const float*)d_in[2];
    const float* wq    = (const float*)d_in[3];
    const float* wk    = (const float*)d_in[4];
    const float* wv    = (const float*)d_in[5];
    const float* wo    = (const float*)d_in[6];
    const float* ln2_w = (const float*)d_in[7];
    const float* wg    = (const float*)d_in[8];
    const float* wu    = (const float*)d_in[9];
    const float* wd    = (const float*)d_in[10];
    const float* lnf_w = (const float*)d_in[11];
    const float* w_out = (const float*)d_in[12];
    float* out = (float*)d_out;

    char* ws = (char*)d_ws;
    size_t off = 0;
    auto carve = [&](size_t bytes) { char* p = ws + off; off += alignup(bytes); return p; };

    float* x0  = (float*)carve((size_t)TT * DD * 4);
    float* x1  = (float*)carve((size_t)TT * DD * 4);
    bf16*  hb  = (bf16*) carve((size_t)TT * DD * 2);
    float* qf  = (float*)carve((size_t)TT * DD * 4);
    float* kf  = (float*)carve((size_t)TT * NKV * DH * 4);
    float* vf  = (float*)carve((size_t)TT * NKV * DH * 4);
    bf16*  qb  = (bf16*) carve((size_t)TT * DD * 2);
    bf16*  kb  = (bf16*) carve((size_t)TT * NKV * DH * 2);
    bf16*  vb  = (bf16*) carve((size_t)TT * NKV * DH * 2);
    bf16*  sab = (bf16*) carve((size_t)TT * DD * 2);
    float* gf  = (float*)carve((size_t)TT * DFF * 4);
    float* uf  = (float*)carve((size_t)TT * DFF * 4);
    bf16*  tb  = (bf16*) carve((size_t)TT * DFF * 2);
    bf16*  wqb = (bf16*) carve((size_t)NL * DD * DD * 2);     // transposed [N][K]
    bf16*  wkb = (bf16*) carve((size_t)NL * DD * NKV * DH * 2);
    bf16*  wvb = (bf16*) carve((size_t)NL * DD * NKV * DH * 2);
    bf16*  wob = (bf16*) carve((size_t)NL * DD * DD * 2);
    bf16*  wgb = (bf16*) carve((size_t)NL * DD * DFF * 2);
    bf16*  wub = (bf16*) carve((size_t)NL * DD * DFF * 2);
    bf16*  wdb = (bf16*) carve((size_t)NL * DFF * DD * 2);
    bf16*  wot = (bf16*) carve((size_t)DD * VV * 2);

    auto castT = [&](const float* s, bf16* d, int Kd, int Nd) {
        cast_transpose_kernel<<<dim3(Nd / 32, Kd / 32), dim3(32, 8), 0, stream>>>(s, d, Kd, Nd);
    };
    for (int l = 0; l < NL; ++l) {
        castT(wq + (size_t)l * DD * DD,       wqb + (size_t)l * DD * DD,       DD,  DD);
        castT(wk + (size_t)l * DD * NKV * DH, wkb + (size_t)l * DD * NKV * DH, DD,  NKV * DH);
        castT(wv + (size_t)l * DD * NKV * DH, wvb + (size_t)l * DD * NKV * DH, DD,  NKV * DH);
        castT(wo + (size_t)l * DD * DD,       wob + (size_t)l * DD * DD,       DD,  DD);
        castT(wg + (size_t)l * DD * DFF,      wgb + (size_t)l * DD * DFF,      DD,  DFF);
        castT(wu + (size_t)l * DD * DFF,      wub + (size_t)l * DD * DFF,      DD,  DFF);
        castT(wd + (size_t)l * DFF * DD,      wdb + (size_t)l * DFF * DD,      DFF, DD);
    }
    castT(w_out, wot, DD, VV);

    embed_kernel<<<dim3(TT), 256, 0, stream>>>(ti, emb, x0);

    auto gemm = [&](const bf16* A, const bf16* Bt, float* C, const float* Res,
                    int M, int N, int K) {
        dim3 grid(N / BN, M / BM);
        gemm_bf16_kernel<<<grid, 256, 0, stream>>>(A, Bt, C, Res, M, N, K);
    };

    for (int l = 0; l < NL; ++l) {
        rmsnorm_kernel<<<dim3(TT), 256, 0, stream>>>(x0, ln1_w + (size_t)l * DD, hb);
        gemm(hb, wqb + (size_t)l * DD * DD,       qf, nullptr, TT, DD,       DD);
        gemm(hb, wkb + (size_t)l * DD * NKV * DH, kf, nullptr, TT, NKV * DH, DD);
        gemm(hb, wvb + (size_t)l * DD * NKV * DH, vf, nullptr, TT, NKV * DH, DD);
        rope_cast_kernel<<<dim3(TT), 256, 0, stream>>>(qf, kf, vf, qb, kb, vb);
        attn_kernel<<<dim3(BB * NH, LL / 16), 32, 0, stream>>>(qb, kb, vb, sab);
        gemm(sab, wob + (size_t)l * DD * DD, x1, x0, TT, DD, DD);   // + residual
        rmsnorm_kernel<<<dim3(TT), 256, 0, stream>>>(x1, ln2_w + (size_t)l * DD, hb);
        gemm(hb, wgb + (size_t)l * DD * DFF, gf, nullptr, TT, DFF, DD);
        gemm(hb, wub + (size_t)l * DD * DFF, uf, nullptr, TT, DFF, DD);
        {
            size_t n = (size_t)TT * DFF;
            silu_mul_kernel<<<dim3((unsigned)((n + 255) / 256)), 256, 0, stream>>>(gf, uf, tb, n);
        }
        gemm(tb, wdb + (size_t)l * DFF * DD, x0, x1, TT, DD, DFF);  // + residual
    }

    rmsnorm_kernel<<<dim3(TT), 256, 0, stream>>>(x0, lnf_w, hb);
    // logits GEMM: A and B tiles DMA'd by the Tensor Data Mover, NT output
    gemm_bf16_tdm_kernel<<<dim3(VV / BN, TT / BM), 256, 0, stream>>>(hb, wot, out,
                                                                     TT, VV, DD);
}
